// WindowAttention_4621384810902
// MI455X (gfx1250) — compile-verified, matrix operand order checked by reading, not run on hardware
//
#include <hip/hip_runtime.h>
#include <math.h>
#include <stdint.h>

typedef __bf16 bf16;
typedef __attribute__((ext_vector_type(16))) bf16 v16bf;
typedef __attribute__((ext_vector_type(8)))  float v8f;
typedef __attribute__((ext_vector_type(4)))  int   v4i;

#define DIM      192
#define HEADS    6
#define HD       32
#define NTOK     64
#define BATCH    2048
#define QKV_COLS 576
#define CPBH     512
#define RPB_N    343

// ---------------------------------------------------------------------------
// WMMA helpers. CDNA5 16-bit A/B VGPR layout (CDNA5 ISA §7.12.2):
// lane L holds row (A: M=L&15, B: N=L&15), half h=L>>4 selects K phase.
// Packed 16-bit slot e (0..15): K = k0 + h*8 + ((e&8)<<1) + (e&7).
// C/D f32: lane -> N=L&15, VGPR r -> M = r + (L>>4)*8.
// ---------------------------------------------------------------------------
__device__ __forceinline__ v8f wmma_bf16(v16bf a, v16bf b, v8f c) {
  return __builtin_amdgcn_wmma_f32_16x16x32_bf16(false, a, false, b,
                                                 (short)0, c, false, false);
}

// 16x32 fragment from an LDS tile stored row-major (K contiguous, ld elems).
__device__ __forceinline__ v16bf frag_ld_lds(const bf16* base, int ld,
                                             int row0, int k0) {
  const int l = threadIdx.x & 31;
  const bf16* p = base + (row0 + (l & 15)) * ld + k0 + ((l >> 4) << 3);
  v16bf f;
#pragma unroll
  for (int e = 0; e < 16; ++e) {
    const int k = ((e & 8) << 1) + (e & 7);
    f[e] = p[k];
  }
  return f;
}

// B fragment for D = A * W^T from a pre-converted global bf16 weight stored
// row-major (rows = output col n, K contiguous). 2 x global_load_b128.
__device__ __forceinline__ v16bf frag_ld_gb(const bf16* __restrict__ W,
                                            int ld, int n0, int k0) {
  const int l = threadIdx.x & 31;
  const bf16* p = W + (size_t)(n0 + (l & 15)) * ld + k0 + ((l >> 4) << 3);
  v16bf f;
#pragma unroll
  for (int e = 0; e < 16; ++e) {
    const int k = ((e & 8) << 1) + (e & 7);
    f[e] = p[k];
  }
  return f;
}

// ---------------------------------------------------------------------------
// One-shot fp32 -> bf16 weight conversion (n multiple of 4).
// ---------------------------------------------------------------------------
__global__ void __launch_bounds__(256) k_cvt(const float* __restrict__ src,
                                             bf16* __restrict__ dst, int n) {
  const int i = (blockIdx.x * 256 + threadIdx.x) * 4;
  if (i < n) {
    const float4 v = *(const float4*)(src + i);
    dst[i + 0] = (bf16)v.x; dst[i + 1] = (bf16)v.y;
    dst[i + 2] = (bf16)v.z; dst[i + 3] = (bf16)v.w;
  }
}

// ---------------------------------------------------------------------------
// Kernel 0: continuous-position-bias MLP + gather + logit scale (tiny).
// biasM[h][i][j] = 16*sigmoid( (relu(rpb @ w1^T + b1) @ w2^T)[rpb_idx[i,j], h] )
// scale[h] = exp(min(logit_scale[h], log(100)))
// ---------------------------------------------------------------------------
__global__ void __launch_bounds__(512) k_bias(
    const float* __restrict__ rpb, const int* __restrict__ rpb_idx,
    const float* __restrict__ w1, const float* __restrict__ b1,
    const float* __restrict__ w2, const float* __restrict__ lscale,
    float* __restrict__ biasM, float* __restrict__ scale) {
  __shared__ float tab[RPB_N * HEADS];
  const int t = threadIdx.x;
  if (t < RPB_N) {
    const float r0 = rpb[t * 3 + 0], r1 = rpb[t * 3 + 1], r2 = rpb[t * 3 + 2];
    float acc[HEADS] = {};
    for (int c = 0; c < CPBH; ++c) {
      float hv = r0 * w1[c * 3 + 0] + r1 * w1[c * 3 + 1] + r2 * w1[c * 3 + 2]
                 + b1[c];
      hv = fmaxf(hv, 0.0f);
#pragma unroll
      for (int h = 0; h < HEADS; ++h) acc[h] += hv * w2[h * CPBH + c];
    }
#pragma unroll
    for (int h = 0; h < HEADS; ++h) tab[t * HEADS + h] = acc[h];
  }
  if (t < HEADS) scale[t] = expf(fminf(lscale[t], logf(100.0f)));
  __syncthreads();
  for (int i = t; i < HEADS * NTOK * NTOK; i += blockDim.x) {
    const int h = i >> 12;        // / 4096
    const int r = i & 4095;       // (i,j) flat
    const float v = tab[rpb_idx[r] * HEADS + h];
    biasM[i] = 16.0f / (1.0f + expf(-v));
  }
}

// ---------------------------------------------------------------------------
// Kernel 1: QKV projection. One block per window (64 rows), 8 waves.
// x tile is staged global->LDS with CDNA5 async-to-LDS ops (ASYNCcnt) when
// available, then converted bf16 in place (aliased buffer).
// C(64x576) = x(64x192) @ qkv_w^T + qkv_b, stored bf16 (cosine normalization
// happens in k_attn where full hd rows are staged).
// ---------------------------------------------------------------------------
__global__ void __launch_bounds__(256) k_qkv(
    const float* __restrict__ x, const bf16* __restrict__ qkv_wb,
    const float* __restrict__ qkv_b, bf16* __restrict__ qkvb) {
  __shared__ alignas(16) float xf[NTOK * DIM];   // 48 KB raw fp32 stage
  bf16* xb = (bf16*)xf;                          // aliased bf16 tile (24 KB)
  const int b = blockIdx.x;
  const int tid = threadIdx.x;
  const int wave = tid >> 5;
  const int l = tid & 31;

  // Warm the (L2-resident) bf16 weight.
  for (int i = tid; i < QKV_COLS; i += 256)
    __builtin_prefetch(qkv_wb + (size_t)i * DIM, 0, 0);

  const float* xw = x + (size_t)b * NTOK * DIM;

#if __has_builtin(__builtin_amdgcn_global_load_async_to_lds_b128)
  {
    // Each thread async-copies the 48 floats it will convert (12 x b128).
    typedef __attribute__((address_space(1))) v4i* gv4p;
    typedef __attribute__((address_space(3))) v4i* lv4p;
    const int base = tid * 48;
#pragma unroll
    for (int j = 0; j < 12; ++j) {
      gv4p gp = (gv4p)(uintptr_t)(xw + base + j * 4);
      lv4p lp = (lv4p)(xf + base + j * 4);
      __builtin_amdgcn_global_load_async_to_lds_b128(gp, lp, 0, 0);
    }
#if __has_builtin(__builtin_amdgcn_s_wait_asynccnt)
    __builtin_amdgcn_s_wait_asynccnt(0);
#else
    asm volatile("s_wait_asynccnt 0" ::: "memory");
#endif
    // Convert own region fp32 -> bf16 into the aliased front of the buffer.
    float tmp[48];
#pragma unroll
    for (int j = 0; j < 48; ++j) tmp[j] = xf[base + j];
    __syncthreads();
#pragma unroll
    for (int j = 0; j < 48; ++j) xb[base + j] = (bf16)tmp[j];
  }
#else
  for (int i = tid * 4; i < NTOK * DIM; i += 256 * 4) {
    const float4 v = *(const float4*)(xw + i);
    xb[i + 0] = (bf16)v.x; xb[i + 1] = (bf16)v.y;
    xb[i + 2] = (bf16)v.z; xb[i + 3] = (bf16)v.w;
  }
#endif
  __syncthreads();

  bf16* outp = qkvb + (size_t)b * NTOK * QKV_COLS;
  // 4 row-tiles x 36 col-tiles = 144 tiles; 18 per wave; K = 192 = 6 steps.
  for (int i = 0; i < 18; ++i) {
    const int t = wave * 18 + i;
    const int m0 = (t & 3) << 4;
    const int n0 = (t >> 2) << 4;
    v8f acc = {};
#pragma unroll
    for (int kk = 0; kk < 6; ++kk) {
      const v16bf a = frag_ld_lds(xb, DIM, m0, kk * 32);
      const v16bf w = frag_ld_gb(qkv_wb, DIM, n0, kk * 32);
      acc = wmma_bf16(a, w, acc);
    }
    const int n = n0 + (l & 15);
    const float bias = qkv_b[n];
#pragma unroll
    for (int r = 0; r < 8; ++r) {
      const int m = m0 + ((l >> 4) << 3) + r;
      outp[m * QKV_COLS + n] = (bf16)(acc[r] + bias);
    }
  }
}

// ---------------------------------------------------------------------------
// Kernel 2: fused cosine attention + output projection. One block per window.
// Per head: normalize q,k (f32), stage v^T; S = qn kn^T * scale + bias + mask
// (one 16x16x32 WMMA per score tile since hd==32); row softmax (4 lanes/row);
// out_h = P @ v via WMMA; heads accumulate into a 64x192 bf16 LDS buffer;
// then proj: out @ proj_w^T + proj_b -> f32 global.
// ---------------------------------------------------------------------------
__global__ void __launch_bounds__(256) k_attn(
    const bf16* __restrict__ qkvb, const float* __restrict__ mask,
    const float* __restrict__ biasM, const float* __restrict__ scaleA,
    const bf16* __restrict__ proj_wb, const float* __restrict__ proj_b,
    float* __restrict__ out) {
  __shared__ alignas(16) bf16 qh[NTOK * HD];       // 4 KB
  __shared__ alignas(16) bf16 kh[NTOK * HD];       // 4 KB
  __shared__ alignas(16) bf16 vT[HD * NTOK];       // 4 KB
  __shared__ alignas(16) float S[NTOK * NTOK];     // 16 KB
  __shared__ alignas(16) bf16 P[NTOK * NTOK];      // 8 KB
  __shared__ alignas(16) bf16 outb[NTOK * DIM];    // 24 KB   (total 60 KB)

  const int b = blockIdx.x;
  const int tid = threadIdx.x;
  const int wave = tid >> 5;
  const int l = tid & 31;
  const bf16* base = qkvb + (size_t)b * NTOK * QKV_COLS;
  const float* maskw = mask + (size_t)(b & 63) * NTOK * NTOK;

  for (int h = 0; h < HEADS; ++h) {
    const float sc = scaleA[h];
    // ---- stage q (normalized), k (normalized), v^T -----------------------
    if (tid < 192) {
      const int which = tid / 64;        // 0=q, 1=k, 2=v
      const int m = tid & 63;
      const bf16* src = base + m * QKV_COLS + which * DIM + h * HD;
      float vals[HD];
#pragma unroll
      for (int d = 0; d < HD; ++d) vals[d] = (float)src[d];
      if (which < 2) {
        float ss = 0.0f;
#pragma unroll
        for (int d = 0; d < HD; ++d) ss += vals[d] * vals[d];
        const float inv = 1.0f / fmaxf(sqrtf(ss), 1e-12f);
        bf16* dst = (which == 0 ? qh : kh) + m * HD;
#pragma unroll
        for (int d = 0; d < HD; ++d) dst[d] = (bf16)(vals[d] * inv);
      } else {
#pragma unroll
        for (int d = 0; d < HD; ++d) vT[d * NTOK + m] = (bf16)vals[d];
      }
    }
    __syncthreads();

    // ---- scores: 16 tiles, 2 per wave, single K=32 WMMA each -------------
#pragma unroll
    for (int i = 0; i < 2; ++i) {
      const int t = wave * 2 + i;
      const int m0 = (t & 3) << 4;
      const int n0 = (t >> 2) << 4;
      const v16bf a  = frag_ld_lds(qh, HD, m0, 0);
      const v16bf bb = frag_ld_lds(kh, HD, n0, 0);
      v8f acc = {};
      acc = wmma_bf16(a, bb, acc);
      const int n = n0 + (l & 15);
#pragma unroll
      for (int r = 0; r < 8; ++r) {
        const int m = m0 + ((l >> 4) << 3) + r;
        S[m * NTOK + n] = acc[r] * sc + biasM[h * 4096 + m * NTOK + n]
                          + maskw[m * NTOK + n];
      }
    }
    __syncthreads();

    // ---- softmax: 4 contiguous lanes per row (shfl_xor within wave) ------
    {
      const int row = tid >> 2, qd = tid & 3;
      const float* sr = S + row * NTOK + qd * 16;
      float mx = -3.0e38f;
#pragma unroll
      for (int j = 0; j < 16; ++j) mx = fmaxf(mx, sr[j]);
      mx = fmaxf(mx, __shfl_xor(mx, 1, 32));
      mx = fmaxf(mx, __shfl_xor(mx, 2, 32));
      float e[16];
      float sm = 0.0f;
#pragma unroll
      for (int j = 0; j < 16; ++j) { e[j] = __expf(sr[j] - mx); sm += e[j]; }
      sm += __shfl_xor(sm, 1, 32);
      sm += __shfl_xor(sm, 2, 32);
      const float inv = 1.0f / sm;
      bf16* pr = P + row * NTOK + qd * 16;
#pragma unroll
      for (int j = 0; j < 16; ++j) pr[j] = (bf16)(e[j] * inv);
    }
    __syncthreads();

    // ---- PV: 8 tiles (4 row x 2 col), one per wave, K = 64 = 2 WMMA ------
    {
      const int m0 = (wave & 3) << 4;
      const int d0 = (wave >> 2) << 4;
      v8f acc = {};
#pragma unroll
      for (int kk = 0; kk < 2; ++kk) {
        const v16bf a  = frag_ld_lds(P,  NTOK, m0, kk * 32);
        const v16bf bb = frag_ld_lds(vT, NTOK, d0, kk * 32);
        acc = wmma_bf16(a, bb, acc);
      }
      const int n = d0 + (l & 15);
#pragma unroll
      for (int r = 0; r < 8; ++r) {
        const int m = m0 + ((l >> 4) << 3) + r;
        outb[m * DIM + h * HD + n] = (bf16)acc[r];
      }
    }
    __syncthreads();
  }

  // ---- output projection: 48 tiles, 6 per wave, K = 192 = 6 WMMA ---------
  float* op = out + (size_t)b * NTOK * DIM;
  for (int i = 0; i < 6; ++i) {
    const int t = wave * 6 + i;
    const int m0 = (t % 4) << 4;
    const int n0 = (t / 4) << 4;
    v8f acc = {};
#pragma unroll
    for (int kk = 0; kk < 6; ++kk) {
      const v16bf a = frag_ld_lds(outb, DIM, m0, kk * 32);
      const v16bf w = frag_ld_gb(proj_wb, DIM, n0, kk * 32);
      acc = wmma_bf16(a, w, acc);
    }
    const int n = n0 + (l & 15);
    const float pb = proj_b[n];
#pragma unroll
    for (int r = 0; r < 8; ++r) {
      const int m = m0 + ((l >> 4) << 3) + r;
      op[m * DIM + n] = acc[r] + pb;
    }
  }
}

// ---------------------------------------------------------------------------
extern "C" void kernel_launch(void* const* d_in, const int* in_sizes, int n_in,
                              void* d_out, int out_size, void* d_ws,
                              size_t ws_size, hipStream_t stream) {
  const float* x       = (const float*)d_in[0];
  const float* mask    = (const float*)d_in[1];
  const float* rpb     = (const float*)d_in[2];
  const int*   rpb_idx = (const int*)  d_in[3];
  const float* cpb_w1  = (const float*)d_in[4];
  const float* cpb_b1  = (const float*)d_in[5];
  const float* cpb_w2  = (const float*)d_in[6];
  const float* qkv_w   = (const float*)d_in[7];
  const float* qkv_b   = (const float*)d_in[8];
  const float* proj_w  = (const float*)d_in[9];
  const float* proj_b  = (const float*)d_in[10];
  const float* lscale  = (const float*)d_in[11];

  char* ws = (char*)d_ws;
  size_t off = 0;
  bf16* qkvb = (bf16*)(ws + off);                               // 151.0 MB
  off += (size_t)BATCH * NTOK * QKV_COLS * sizeof(bf16);
  float* biasM = (float*)(ws + off);                            // 96 KB
  off += (size_t)HEADS * NTOK * NTOK * sizeof(float);
  float* scale = (float*)(ws + off);                            // 64 B
  off += 64;
  bf16* qkv_wb = (bf16*)(ws + off);                             // 216 KB
  off += (size_t)QKV_COLS * DIM * sizeof(bf16);
  bf16* proj_wb = (bf16*)(ws + off);                            // 72 KB
  off += (size_t)DIM * DIM * sizeof(bf16);

  k_cvt<<<(QKV_COLS * DIM / 4 + 255) / 256, 256, 0, stream>>>(
      qkv_w, qkv_wb, QKV_COLS * DIM);
  k_cvt<<<(DIM * DIM / 4 + 255) / 256, 256, 0, stream>>>(
      proj_w, proj_wb, DIM * DIM);
  k_bias<<<1, 512, 0, stream>>>(rpb, rpb_idx, cpb_w1, cpb_b1, cpb_w2, lscale,
                                biasM, scale);
  k_qkv<<<BATCH, 256, 0, stream>>>(x, qkv_wb, qkv_b, qkvb);
  k_attn<<<BATCH, 256, 0, stream>>>(qkvb, mask, biasM, scale, proj_wb, proj_b,
                                    (float*)d_out);
  (void)in_sizes; (void)n_in; (void)out_size; (void)ws_size;
}
